// LiquidODECell_11957188952457
// MI455X (gfx1250) — compile-verified, matrix operand order checked by reading, not run on hardware
//
#include <hip/hip_runtime.h>
#include <hip/hip_bf16.h>

// ---------------- Types ----------------
typedef __bf16 bf16;
typedef __attribute__((ext_vector_type(16))) __bf16 v16bf;
typedef __attribute__((ext_vector_type(8)))  float  v8f;
typedef __attribute__((ext_vector_type(4)))  unsigned int u32x4;
typedef __attribute__((ext_vector_type(8)))  unsigned int u32x8;

// ---------------- Problem constants ----------------
#define BDIM    32768
#define DIN     256
#define HDIM    256
#define STEPS   3
#define DT      (1.0f / 3.0f)
#define TAU_MIN 0.2f
#define ALPHA   0.1f
#define ETA     0.1f
#define DECAY   0.99f
#define MOE_W   1.0f

// dyn kernel tiling
#define BMROW 32          // batch rows per workgroup
#define LDP   264         // padded LDS row stride (bf16 elems); TDM pad reproduces this

// hebb kernel tiling
#define KC    32
#define HT    64
#define HP    72
#define NSPLIT 16

static __device__ inline v8f wmma_bf16(v16bf a, v16bf b, v8f c) {
    return __builtin_amdgcn_wmma_f32_16x16x32_bf16(
        /*neg_a=*/false, a, /*neg_b=*/false, b,
        /*c_mod=*/(short)0, c, /*reuse_a=*/false, /*reuse_b=*/false);
}

// A fragment (16x32, row-major tile in LDS). lane 0-15: M=lane, K=0..7,16..23 ; lane16-31: K=8..15,24..31
static __device__ inline v16bf loadA(const bf16* tile, int mbase, int kof, int lane) {
    int m  = mbase + (lane & 15);
    int k0 = kof + ((lane >> 4) << 3);
    const bf16* p = tile + m * LDP + k0;
    v16bf a;
#pragma unroll
    for (int i = 0; i < 8; ++i) { a[i] = p[i]; a[i + 8] = p[i + 16]; }
    return a;
}

// B fragment from pre-swizzled global weight buffer: per (kb,nb) block, 32 lanes x 16 contiguous bf16
static __device__ inline v16bf loadB(const bf16* frag, int kb, int nb, int nblocks, int lane) {
    const bf16* p = frag + (((size_t)(kb * nblocks + nb)) * 32 + lane) * 16;
    v16bf b;
#pragma unroll
    for (int i = 0; i < 16; ++i) b[i] = p[i];
    return b;
}

// ---------------- TDM: DMA a 32-row x 512B tile (bf16 256/row) into LDS with 16B/row padding ----
// Descriptor per CDNA5 ISA. data_size=4B, tile 128(dw) x 32, tensor row
// stride = 128 dw, pad 4 dwords (code 3) after every 512 bytes (interval code 6) -> LDS stride 528B = LDP.
static __device__ inline void tdm_load_tile(unsigned long long gaddr, unsigned int ldsaddr) {
    u32x4 g0;
    g0[0] = 1u;                                                  // count=1, user descriptor
    g0[1] = ldsaddr;                                             // lds_addr (bytes)
    g0[2] = (unsigned int)gaddr;                                 // global_addr[31:0]
    g0[3] = (unsigned int)((gaddr >> 32) & 0x1FFFFFFu)           // global_addr[56:32]
          | (2u << 30);                                          // type = 2 (image)
    u32x8 g1;
    g1[0] = (2u << 16)        // data_size = 4 bytes
          | (1u << 20)        // pad_enable
          | (6u << 22)        // pad_interval: 64 x 8B = 512B
          | (3u << 25);       // pad_amount: 4 dwords = 16B
    g1[1] = (128u << 16);     // tensor_dim0 = 128 (dwords per row)
    g1[2] = (32768u << 16);   // tensor_dim1[15:0] = 32768 rows
    g1[3] = (128u << 16);     // tile_dim0 = 128 dwords
    g1[4] = 32u;              // tile_dim1 = 32 rows
    g1[5] = 128u;             // tensor_dim0_stride = 128 dwords
    g1[6] = 0u;
    g1[7] = 0u;
    asm volatile("tensor_load_to_lds %0, %1" :: "s"(g0), "s"(g1) : "memory");
}

// ---------------- Elementwise helpers ----------------
__global__ __launch_bounds__(256) void k_cvt_bf16(const float* __restrict__ s, bf16* __restrict__ d, int n) {
    for (int i = blockIdx.x * 256 + threadIdx.x; i < n; i += gridDim.x * 256) d[i] = (bf16)s[i];
}
__global__ __launch_bounds__(256) void k_copy_cvt(const float* __restrict__ s, float* __restrict__ df,
                                                  bf16* __restrict__ db, int n) {
    for (int i = blockIdx.x * 256 + threadIdx.x; i < n; i += gridDim.x * 256) {
        float v = s[i]; df[i] = v; db[i] = (bf16)v;
    }
}
__global__ __launch_bounds__(256) void k_copy(const float* __restrict__ s, float* __restrict__ d, int n) {
    for (int i = blockIdx.x * 256 + threadIdx.x; i < n; i += gridDim.x * 256) d[i] = s[i];
}
__global__ __launch_bounds__(256) void k_scale(float* __restrict__ p, float f, int n) {
    for (int i = blockIdx.x * 256 + threadIdx.x; i < n; i += gridDim.x * 256) p[i] *= f;
}

// Build B-matrix fragment buffer: B[k][n] = W[n*K + k] (+ alpha*hebb[k*N + n])
__global__ __launch_bounds__(256) void k_build_fragB(const float* __restrict__ W,
                                                     const float* __restrict__ hebb, float alpha,
                                                     bf16* __restrict__ frag, int K, int N) {
    int idx = blockIdx.x * 256 + threadIdx.x;
    if (idx >= K * N) return;
    int k = idx / N, n = idx - k * N;
    float v = W[n * K + k];
    if (hebb) v += alpha * hebb[k * N + n];
    int kb = k >> 5, kin = k & 31, nb = n >> 4, nin = n & 15;
    int hi = kin >> 3;
    int e  = (kin & 7) | ((hi >> 1) << 3);
    int lan = nin + ((hi & 1) << 4);
    frag[(((size_t)(kb * (N >> 4) + nb)) * 32 + lan) * 16 + e] = (bf16)v;
}

// ---------------- Fused dynamics kernel ----------------
// out = h_base + scale * ( (-h_eval + tanh(x@Wih + act@Whh + b)) / tau(x,act) )
__global__ __launch_bounds__(256) void dyn_kernel(
    const bf16* __restrict__ xb, const bf16* __restrict__ actb,
    const float* __restrict__ h_eval, const float* __restrict__ h_base,
    const bf16* __restrict__ Wihf, const bf16* __restrict__ Whhf,
    const bf16* __restrict__ Wt1f, const bf16* __restrict__ Wt2f,
    const float* __restrict__ b_ih, const float* __restrict__ b_hh,
    const float* __restrict__ b_t1, const float* __restrict__ b_t2,
    float* __restrict__ outf, bf16* __restrict__ outb, float scale)
{
    __shared__ bf16 sMem[3 * BMROW * LDP];
    bf16* sX = sMem;
    bf16* sA = sMem + BMROW * LDP;
    bf16* sU = sMem + 2 * BMROW * LDP;

    const int t = threadIdx.x;
    const int lane = t & 31, wid = t >> 5;
    const int rowbase = blockIdx.x * BMROW;

    // ---- async tensor DMA of x-tile and act-tile into LDS (wave 0 issues; all waves barrier) ----
    if (wid == 0) {
        tdm_load_tile((unsigned long long)(uintptr_t)(xb + (size_t)rowbase * HDIM),
                      (unsigned int)(unsigned long long)(uintptr_t)(void*)sX);
        tdm_load_tile((unsigned long long)(uintptr_t)(actb + (size_t)rowbase * HDIM),
                      (unsigned int)(unsigned long long)(uintptr_t)(void*)sA);
        __builtin_amdgcn_s_wait_tensorcnt(0);
    }
    __syncthreads();

    const int mw = wid >> 2;      // 0..1
    const int nw = wid & 3;       // 0..3
    const int mbase = mw * 16;
    const int nbase = nw * 64;
    const int lhalf = lane >> 4;
    const int lmod  = lane & 15;

    // ---- phase A: u = silu([x|act] @ W_t1^T + b_t1), K = 512 ----
    v8f accU[4];
#pragma unroll
    for (int nt = 0; nt < 4; ++nt) accU[nt] = (v8f){0,0,0,0,0,0,0,0};
#pragma unroll 1
    for (int kb = 0; kb < 16; ++kb) {
        const bf16* src = (kb < 8) ? sX : sA;
        v16bf a = loadA(src, mbase, (kb & 7) * 32, lane);
#pragma unroll
        for (int nt = 0; nt < 4; ++nt) {
            v16bf b = loadB(Wt1f, kb, (nw << 2) + nt, 16, lane);
            accU[nt] = wmma_bf16(a, b, accU[nt]);
        }
    }
#pragma unroll
    for (int nt = 0; nt < 4; ++nt) {
        int col = nbase + nt * 16 + lmod;
        float bt = b_t1[col];
#pragma unroll
        for (int r = 0; r < 8; ++r) {
            float u = accU[nt][r] + bt;
            float s = u / (1.0f + __expf(-u));     // SiLU
            sU[(mbase + r + 8 * lhalf) * LDP + col] = (bf16)s;
        }
    }
    __syncthreads();

    // ---- phase B: tau = softplus(u @ W_t2^T + b_t2) + TAU_MIN, K = 256 ----
    v8f accT[4];
#pragma unroll
    for (int nt = 0; nt < 4; ++nt) accT[nt] = (v8f){0,0,0,0,0,0,0,0};
#pragma unroll 1
    for (int kb = 0; kb < 8; ++kb) {
        v16bf a = loadA(sU, mbase, kb * 32, lane);
#pragma unroll
        for (int nt = 0; nt < 4; ++nt) {
            v16bf b = loadB(Wt2f, kb, (nw << 2) + nt, 16, lane);
            accT[nt] = wmma_bf16(a, b, accT[nt]);
        }
    }
    float tauv[4][8];
#pragma unroll
    for (int nt = 0; nt < 4; ++nt) {
        int col = nbase + nt * 16 + lmod;
        float bt = b_t2[col];
#pragma unroll
        for (int r = 0; r < 8; ++r) {
            float v = accT[nt][r] + bt;
            float sp = (v > 20.0f) ? v : log1pf(__expf(v));
            tauv[nt][r] = sp + TAU_MIN;
        }
    }

    // ---- phase C: interaction = tanh(x@Wih + act@Whh + b_ih + b_hh), K = 256 + 256 ----
    v8f accI[4];
#pragma unroll
    for (int nt = 0; nt < 4; ++nt) accI[nt] = (v8f){0,0,0,0,0,0,0,0};
#pragma unroll 1
    for (int kb = 0; kb < 8; ++kb) {
        v16bf a = loadA(sX, mbase, kb * 32, lane);
#pragma unroll
        for (int nt = 0; nt < 4; ++nt) {
            v16bf b = loadB(Wihf, kb, (nw << 2) + nt, 16, lane);
            accI[nt] = wmma_bf16(a, b, accI[nt]);
        }
    }
#pragma unroll 1
    for (int kb = 0; kb < 8; ++kb) {
        v16bf a = loadA(sA, mbase, kb * 32, lane);
#pragma unroll
        for (int nt = 0; nt < 4; ++nt) {
            v16bf b = loadB(Whhf, kb, (nw << 2) + nt, 16, lane);
            accI[nt] = wmma_bf16(a, b, accI[nt]);
        }
    }

#pragma unroll
    for (int nt = 0; nt < 4; ++nt) {
        int col = nbase + nt * 16 + lmod;
        float bb = b_ih[col] + b_hh[col];
#pragma unroll
        for (int r = 0; r < 8; ++r) {
            int row = rowbase + mbase + r + 8 * lhalf;
            size_t idx = (size_t)row * HDIM + col;
            float tn = tanhf(accI[nt][r] + bb);
            float k  = (-h_eval[idx] + tn) / tauv[nt][r];
            float o  = h_base[idx] + scale * k;
            outf[idx] = o;
            if (outb) outb[idx] = (bf16)o;
        }
    }
}

// ---------------- Hebbian outer-product accumulate (split-K + atomics) ----------------
// dst[p][q] += c * sum_b pre[b][p] * post[b][q]
__global__ __launch_bounds__(256) void hebb_acc(const bf16* __restrict__ pre, const bf16* __restrict__ post,
                                                float* __restrict__ dst, float c, int kPerSplit)
{
    __shared__ bf16 sP[KC * HP];
    __shared__ bf16 sQ[KC * HP];
    const int t = threadIdx.x, lane = t & 31, wid = t >> 5;
    const int tileid = blockIdx.x & 15;
    const int split  = blockIdx.x >> 4;
    const int tm = (tileid >> 2) * HT;
    const int tn = (tileid & 3) * HT;
    const int kstart = split * kPerSplit;
    const int mw = wid >> 1, nw = wid & 1;
    const int lmod = lane & 15, khalf = (lane >> 4) * 8;

    v8f acc[2];
    acc[0] = (v8f){0,0,0,0,0,0,0,0};
    acc[1] = (v8f){0,0,0,0,0,0,0,0};

#pragma unroll 1
    for (int kk = 0; kk < kPerSplit; kk += KC) {
        for (int i = t; i < KC * HT; i += 256) {
            int r = i >> 6, cc = i & 63;
            size_t b = (size_t)(kstart + kk + r) * HDIM;
            sP[r * HP + cc] = pre[b + tm + cc];
            sQ[r * HP + cc] = post[b + tn + cc];
        }
        __syncthreads();

        int m = mw * 16 + lmod;
        v16bf a;
#pragma unroll
        for (int i = 0; i < 8; ++i) { a[i] = sP[(khalf + i) * HP + m]; a[i + 8] = sP[(khalf + 16 + i) * HP + m]; }
#pragma unroll
        for (int nt = 0; nt < 2; ++nt) {
            int n = nw * 32 + nt * 16 + lmod;
            v16bf b;
#pragma unroll
            for (int i = 0; i < 8; ++i) { b[i] = sQ[(khalf + i) * HP + n]; b[i + 8] = sQ[(khalf + 16 + i) * HP + n]; }
            acc[nt] = wmma_bf16(a, b, acc[nt]);
        }
        __syncthreads();
    }

#pragma unroll
    for (int nt = 0; nt < 2; ++nt)
#pragma unroll
        for (int r = 0; r < 8; ++r) {
            int p = tm + mw * 16 + r + 8 * (lane >> 4);
            int q = tn + nw * 32 + nt * 16 + lmod;
            atomicAdd(&dst[p * HDIM + q], c * acc[nt][r]);
        }
}

// ---------------- Host orchestration ----------------
extern "C" void kernel_launch(void* const* d_in, const int* in_sizes, int n_in,
                              void* d_out, int out_size, void* d_ws, size_t ws_size,
                              hipStream_t stream) {
    const float* x      = (const float*)d_in[0];
    const float* h      = (const float*)d_in[1];
    const float* hebb_ih= (const float*)d_in[2];
    const float* hebb_hh= (const float*)d_in[3];
    const float* W_ih   = (const float*)d_in[4];
    const float* b_ih   = (const float*)d_in[5];
    const float* W_hh   = (const float*)d_in[6];
    const float* b_hh   = (const float*)d_in[7];
    const float* W_t1   = (const float*)d_in[8];
    const float* b_t1   = (const float*)d_in[9];
    const float* W_t2   = (const float*)d_in[10];
    const float* b_t2   = (const float*)d_in[11];
    float* hState = (float*)d_out;             // current h lives in d_out

    char* ws = (char*)d_ws;
    size_t off = 0;
    bf16* xb      = (bf16*)(ws + off); off += (size_t)BDIM * DIN * 2;     // 16 MB
    bf16* hb      = (bf16*)(ws + off); off += (size_t)BDIM * HDIM * 2;    // 16 MB
    bf16* hmidb   = (bf16*)(ws + off); off += (size_t)BDIM * HDIM * 2;    // 16 MB
    float* hmid   = (float*)(ws + off); off += (size_t)BDIM * HDIM * 4;   // 32 MB
    float* hebbIw = (float*)(ws + off); off += (size_t)DIN * HDIM * 4;    // 256 KB (keep ih,hh adjacent)
    float* hebbHw = (float*)(ws + off); off += (size_t)HDIM * HDIM * 4;
    bf16* Wihf    = (bf16*)(ws + off); off += (size_t)DIN * HDIM * 2;
    bf16* Whhf    = (bf16*)(ws + off); off += (size_t)HDIM * HDIM * 2;
    bf16* Wt1f    = (bf16*)(ws + off); off += (size_t)(DIN + HDIM) * HDIM * 2;
    bf16* Wt2f    = (bf16*)(ws + off); off += (size_t)HDIM * HDIM * 2;
    (void)off; (void)ws_size; (void)in_sizes; (void)n_in; (void)out_size;

    const int nBH = BDIM * HDIM;               // 8,388,608
    const int ebl = 4096;                      // blocks for elementwise (grid-stride)

    // ---- setup ----
    k_cvt_bf16<<<ebl, 256, 0, stream>>>(x, xb, nBH);
    k_copy_cvt<<<ebl, 256, 0, stream>>>(h, hState, hb, nBH);
    k_copy<<<(DIN * HDIM + 255) / 256, 256, 0, stream>>>(hebb_ih, hebbIw, DIN * HDIM);
    k_copy<<<(HDIM * HDIM + 255) / 256, 256, 0, stream>>>(hebb_hh, hebbHw, HDIM * HDIM);
    k_build_fragB<<<((DIN + HDIM) * HDIM + 255) / 256, 256, 0, stream>>>(W_t1, nullptr, 0.0f, Wt1f, DIN + HDIM, HDIM);
    k_build_fragB<<<(HDIM * HDIM + 255) / 256, 256, 0, stream>>>(W_t2, nullptr, 0.0f, Wt2f, HDIM, HDIM);

    const float cHebb = ETA * (MOE_W / (float)STEPS) / (float)BDIM;
    const int dynGrid = BDIM / BMROW;          // 1024
    const int hebbGrid = 16 * NSPLIT;          // 256
    const int kPerSplit = BDIM / NSPLIT;       // 2048

    for (int s = 0; s < STEPS; ++s) {
        // effective weights for k1
        k_build_fragB<<<(DIN * HDIM + 255) / 256, 256, 0, stream>>>(W_ih, hebbIw, ALPHA, Wihf, DIN, HDIM);
        k_build_fragB<<<(HDIM * HDIM + 255) / 256, 256, 0, stream>>>(W_hh, hebbHw, ALPHA, Whhf, HDIM, HDIM);
        // h_mid = h + 0.5*DT*k1
        dyn_kernel<<<dynGrid, 256, 0, stream>>>(xb, hb, hState, hState, Wihf, Whhf, Wt1f, Wt2f,
                                                b_ih, b_hh, b_t1, b_t2, hmid, hmidb, 0.5f * DT);
        // hebb decay + accumulate outer products
        k_scale<<<(2 * DIN * HDIM + 255) / 256, 256, 0, stream>>>(hebbIw, DECAY, 2 * DIN * HDIM);
        hebb_acc<<<hebbGrid, 256, 0, stream>>>(xb,    hmidb, hebbIw, cHebb, kPerSplit);
        hebb_acc<<<hebbGrid, 256, 0, stream>>>(hmidb, hmidb, hebbHw, cHebb, kPerSplit);
        // effective weights for k2
        k_build_fragB<<<(DIN * HDIM + 255) / 256, 256, 0, stream>>>(W_ih, hebbIw, ALPHA, Wihf, DIN, HDIM);
        k_build_fragB<<<(HDIM * HDIM + 255) / 256, 256, 0, stream>>>(W_hh, hebbHw, ALPHA, Whhf, HDIM, HDIM);
        // h = h + DT*k2 (evaluated at h_mid); write bf16 copy for next step
        dyn_kernel<<<dynGrid, 256, 0, stream>>>(xb, hmidb, hmid, hState, Wihf, Whhf, Wt1f, Wt2f,
                                                b_ih, b_hh, b_t1, b_t2, hState, hb, DT);
    }
}
